// DA_conv_82394652607064
// MI455X (gfx1250) — matrix operand order repaired and checked by reading, not compiled, not run on hardware
//
#include <hip/hip_runtime.h>
#include <hip/hip_bf16.h>

// ---------------------------------------------------------------------------
// Fused DDF + dual-attention block for MI455X (gfx1250, wave32).
// Single pass over the 67MB tensors (memory-bound: ~220MB HBM traffic);
// both per-pixel 64x64 1x1 convs run on v_wmma_f32_16x16x32_bf16
// (f32 accumulate), elementwise/residual math in f32 from LDS, and the x0
// halo tile is staged with gfx1250 async global->LDS copies (ASYNCcnt).
// ---------------------------------------------------------------------------

typedef __attribute__((ext_vector_type(16))) __bf16 v16bf;
typedef __attribute__((ext_vector_type(8)))  float  v8f;

__device__ __forceinline__ float leaky01(float v) { return v > 0.f ? v : 0.1f * v; }

// ---------------- dynamic LDS carve (bytes) ----------------
// sm_x0   : [64][326] f32   x0 halo tile (18x18 + row pad)       83,456
// sm_a    : [256][72] bf16  WMMA A operand (x2 tile, then ddf)   36,864
// sm_big  : 17408 f32       h1 [256][68], reused as yout [64][272] 69,632
// sm_mxav : [2][324] f32    channel max / mean over halo          2,592
// sm_kc   : [64][9]  f32    per-channel dynamic kernels           2,304
// sm_wks2 : [64][9]  f32    Wks2                                  2,304
// sm_sa   : [256]    f32    spatial attention                     1,024
// sm_misc : 228      f32    att/bconv/bks1/bks2/wsa/bsa             912
constexpr int SM_X0   = 0;
constexpr int SM_A    = 83456;
constexpr int SM_BIG  = 120320;
constexpr int SM_MXAV = 189952;
constexpr int SM_KC   = 192544;
constexpr int SM_WKS2 = 194848;
constexpr int SM_SA   = 197152;
constexpr int SM_MISC = 198176;
constexpr int SM_TOTAL= 199088;   // < 320KB WGP LDS

// ---------------------------------------------------------------------------
// Kernel 1: per-batch kernel-MLP (kc) and channel attention (att).
// ---------------------------------------------------------------------------
__global__ __launch_bounds__(64) void prep_small(
    const float* __restrict__ x1,  const float* __restrict__ Wkc1,
    const float* __restrict__ Wkc2,const float* __restrict__ Wca1,
    const float* __restrict__ Wca2,
    float* __restrict__ kcg, float* __restrict__ attg)
{
    __shared__ float sx1[64], sh[64], sh2[8];
    const int b = blockIdx.x, t = threadIdx.x;
    sx1[t] = x1[b * 64 + t];
    __syncthreads();
    float a = 0.f;
    for (int c = 0; c < 64; ++c) a += sx1[c] * Wkc1[c * 64 + t];
    sh[t] = leaky01(a);
    if (t < 8) {
        float s = 0.f;
        for (int c = 0; c < 64; ++c) s += sx1[c] * Wca1[c * 8 + t];
        sh2[t] = leaky01(s);
    }
    __syncthreads();
    for (int q = 0; q < 9; ++q) {
        float s = 0.f;
        for (int c = 0; c < 64; ++c) s += sh[c] * Wkc2[c * 576 + t * 9 + q];
        kcg[b * 576 + t * 9 + q] = s;
    }
    float s2 = 0.f;
    for (int j = 0; j < 8; ++j) s2 += sh2[j] * Wca2[j * 64 + t];
    attg[b * 64 + t] = 1.f / (1.f + __expf(-s2));
}

// ---------------------------------------------------------------------------
// Kernel 2: transpose Wks1/Wconv (stored [in][out]) to bf16 [out][in] so the
// WMMA B fragment (N per lane, K contiguous) is a pair of 16B loads.
// ---------------------------------------------------------------------------
__global__ __launch_bounds__(128) void pack_weights(
    const float* __restrict__ Wks1, const float* __restrict__ Wconv,
    __bf16* __restrict__ wt1, __bf16* __restrict__ wt2)
{
    const int i = blockIdx.x * 128 + threadIdx.x;   // 4096 total
    const int o = i >> 6, c = i & 63;
    wt1[i] = (__bf16)Wks1[c * 64 + o];
    wt2[i] = (__bf16)Wconv[c * 64 + o];
}

// ---------------------------------------------------------------------------
// Fragment loaders (wave32), contiguous k order shared by A and B.
// A (16x32 bf16): lane = h*16+m; elems 0..15 = K(kbase+16h ..) contiguous.
// Rows of sm_a are [pix][72] bf16 -> two 16B LDS loads.
// B (32x16 bf16): lane carries N, elems carry K = kbase+16h.. from bf16
// [out][in] weight rows -> two 16B global loads (hot in L2 across all WGs).
// ---------------------------------------------------------------------------
__device__ __forceinline__ v16bf load_a_frag(const __bf16* sm_a, int pix_row,
                                             int ks, int h)
{
    const __bf16* row = sm_a + pix_row * 72 + ks * 32 + 16 * h;
    v16bf a;
    ((float4*)&a)[0] = *(const float4*)(row);
    ((float4*)&a)[1] = *(const float4*)(row + 8);
    return a;
}

__device__ __forceinline__ v16bf load_b_frag(const __bf16* wt, int n,
                                             int ks, int h)
{
    const __bf16* row = wt + n * 64 + ks * 32 + 16 * h;
    v16bf bf;
    ((float4*)&bf)[0] = ((const float4*)row)[0];
    ((float4*)&bf)[1] = ((const float4*)row)[1];
    return bf;
}

// ---------------------------------------------------------------------------
// Kernel 3: fused main pass. One workgroup = one (batch, 16x16 tile).
// 256 threads = 8 wave32; wave w owns pixel M-tiles 2w, 2w+1.
// ---------------------------------------------------------------------------
__global__ __launch_bounds__(256) void ddf_main(
    const float*  __restrict__ x0g,  const float* __restrict__ x2g,
    const float*  __restrict__ bks1, const float* __restrict__ Wks2,
    const float*  __restrict__ bks2, const float* __restrict__ bconv,
    const float*  __restrict__ Wsa,  const float* __restrict__ bsa,
    const float*  __restrict__ kcg,  const float* __restrict__ attg,
    const __bf16* __restrict__ wt_ks1, const __bf16* __restrict__ wt_conv,
    float* __restrict__ outg)
{
    extern __shared__ char smem[];
    float*  sm_x0   = (float*)(smem + SM_X0);
    __bf16* sm_a    = (__bf16*)(smem + SM_A);
    float*  sm_big  = (float*)(smem + SM_BIG);
    float*  sm_mxav = (float*)(smem + SM_MXAV);
    float*  sm_kc   = (float*)(smem + SM_KC);
    float*  sm_wks2 = (float*)(smem + SM_WKS2);
    float*  sm_sa   = (float*)(smem + SM_SA);
    float*  sm_misc = (float*)(smem + SM_MISC);
    float*  sm_att  = sm_misc;        // 64
    float*  sm_bc   = sm_misc + 64;   // 64
    float*  sm_bk1  = sm_misc + 128;  // 64
    float*  sm_bk2  = sm_misc + 192;  // 16
    float*  sm_wsa  = sm_misc + 208;  // 18 + bsa

    const int tid  = threadIdx.x;
    const int lane = tid & 31;        // wave32
    const int wv   = tid >> 5;
    const int h    = lane >> 4;       // half-wave select in WMMA layouts
    const int ml   = lane & 15;
    const int b    = blockIdx.z;
    const int y0   = blockIdx.y * 16;
    const int xb   = blockIdx.x * 16;
    const int px   = tid & 15, py = tid >> 4;   // this thread's pixel

    // ---- stage small parameters ----
    if (tid < 64) {
        sm_att[tid] = attg[b * 64 + tid];
        sm_bc[tid]  = bconv[tid];
        sm_bk1[tid] = bks1[tid];
    }
    if (tid < 9)   sm_bk2[tid] = bks2[tid];
    if (tid < 18)  sm_wsa[tid] = Wsa[tid];
    if (tid == 18) sm_wsa[18]  = bsa[0];
    for (int i = tid; i < 576; i += 256) {
        sm_kc[i]   = kcg[b * 576 + i];
        sm_wks2[i] = Wks2[i];
    }

    // ---- x0 halo (18x18) -> f32 [ch][326] ----
    // Interior tiles: whole halo is in-bounds -> gfx1250 async global->LDS
    // copies (no VGPR round trip, tracked with ASYNCcnt, overlaps the x2
    // staging below). Border tiles: predicated scalar path with zero-fill.
    const bool interior = (blockIdx.x > 0u) & (blockIdx.x < 7u) &
                          (blockIdx.y > 0u) & (blockIdx.y < 7u);
    if (interior) {
        for (int i = tid; i < 64 * 324; i += 256) {
            const int c = i / 324, pos = i - c * 324;
            const int gy = y0 - 1 + pos / 18;
            const int gx = xb - 1 + pos % 18;
            const unsigned lds_off =
                (unsigned)(SM_X0 + (c * 326 + pos) * 4);
            const unsigned long long ga = (unsigned long long)(uintptr_t)
                (x0g + ((size_t)(b * 64 + c) * 128 + gy) * 128 + gx);
            asm volatile("global_load_async_to_lds_b32 %0, %1, off"
                         :: "v"(lds_off), "v"(ga) : "memory");
        }
    } else {
        for (int i = tid; i < 64 * 324; i += 256) {
            const int c = i / 324, pos = i - c * 324;
            const int gy = y0 - 1 + pos / 18;
            const int gx = xb - 1 + pos % 18;
            float v = 0.f;
            if ((unsigned)gy < 128u && (unsigned)gx < 128u)
                v = x0g[((size_t)(b * 64 + c) * 128 + gy) * 128 + gx];
            sm_x0[c * 326 + pos] = v;
        }
    }

    // ---- x2 tile -> bf16 A operand [pix][72] (overlaps async halo copy) ----
    {
        const float* src = x2g + (size_t)b * 64 * 16384
                               + (size_t)(y0 + py) * 128 + (xb + px);
        for (int c = 0; c < 64; ++c)
            sm_a[tid * 72 + c] = (__bf16)src[(size_t)c * 16384];
    }

    if (interior)
        asm volatile("s_wait_asynccnt 0x0" ::: "memory");
    __syncthreads();

    // ================= GEMM1: h1 = x2tile @ Wks1 (WMMA bf16) =================
    const v8f vzero = {0.f, 0.f, 0.f, 0.f, 0.f, 0.f, 0.f, 0.f};
    v8f acc[2][4];
#pragma unroll
    for (int mi = 0; mi < 2; ++mi)
#pragma unroll
        for (int nt = 0; nt < 4; ++nt) acc[mi][nt] = vzero;

#pragma unroll
    for (int ks = 0; ks < 2; ++ks) {
        v16bf a0 = load_a_frag(sm_a, (wv * 2 + 0) * 16 + ml, ks, h);
        v16bf a1 = load_a_frag(sm_a, (wv * 2 + 1) * 16 + ml, ks, h);
#pragma unroll
        for (int nt = 0; nt < 4; ++nt) {
            v16bf bf = load_b_frag(wt_ks1, nt * 16 + ml, ks, h);
            acc[0][nt] = __builtin_amdgcn_wmma_f32_16x16x32_bf16(
                false, a0, false, bf, (short)0, acc[0][nt], false, false);
            acc[1][nt] = __builtin_amdgcn_wmma_f32_16x16x32_bf16(
                false, a1, false, bf, (short)0, acc[1][nt], false, false);
        }
    }
    // h1 = leaky(acc + bks1) -> sm_big [pix][68]  (C-layout: M=r+8h, N=ml)
#pragma unroll
    for (int mi = 0; mi < 2; ++mi)
#pragma unroll
        for (int nt = 0; nt < 4; ++nt)
#pragma unroll
            for (int r = 0; r < 8; ++r) {
                const int p = (wv * 2 + mi) * 16 + r + 8 * h;
                const int c = nt * 16 + ml;
                sm_big[p * 68 + c] = leaky01(acc[mi][nt][r] + sm_bk1[c]);
            }
    __syncthreads();

    // ---- ksp[9] per pixel (64x9 dense, VALU from LDS) ----
    float kspr[9];
    {
        const float* hrow = sm_big + tid * 68;
#pragma unroll
        for (int t = 0; t < 9; ++t) kspr[t] = sm_bk2[t];
        for (int c = 0; c < 64; ++c) {
            const float hv = hrow[c];
#pragma unroll
            for (int t = 0; t < 9; ++t) kspr[t] += hv * sm_wks2[c * 9 + t];
        }
    }

    // ---- channel max/mean over halo positions (spatial attention input) ----
    for (int pos = tid; pos < 324; pos += 256) {
        float mx = -1e30f, sum = 0.f;
        for (int c = 0; c < 64; ++c) {
            const float v = sm_x0[c * 326 + pos];
            mx = fmaxf(mx, v); sum += v;
        }
        sm_mxav[pos]       = mx;
        sm_mxav[324 + pos] = sum * 0.015625f;
    }
    __syncthreads();

    const int pos0 = (py + 1) * 18 + (px + 1);

    // ---- spatial attention: 2->1 3x3 conv + sigmoid ----
    {
        float s = sm_wsa[18];
#pragma unroll
        for (int t = 0; t < 9; ++t) {
            const int d = (t / 3 - 1) * 18 + (t % 3 - 1);
            s += sm_wsa[t]     * sm_mxav[pos0 + d]
               + sm_wsa[9 + t] * sm_mxav[324 + pos0 + d];
        }
        sm_sa[tid] = 1.f / (1.f + __expf(-s));
    }

    // ---- ddf: 9-tap dynamic filter, overwrite sm_a as next A operand ----
#pragma unroll 4
    for (int c = 0; c < 64; ++c) {
        const float* xr  = sm_x0 + c * 326 + pos0;
        const float* kcc = sm_kc + c * 9;
        float s = 0.f;
#pragma unroll
        for (int t = 0; t < 9; ++t) {
            const int d = (t / 3 - 1) * 18 + (t % 3 - 1);
            s += xr[d] * kcc[t] * kspr[t];
        }
        sm_a[tid * 72 + c] = (__bf16)s;
    }
    __syncthreads();

    // ================= GEMM2: y = ddf @ Wconv (WMMA bf16) =================
    v8f acc2[2][4];
#pragma unroll
    for (int mi = 0; mi < 2; ++mi)
#pragma unroll
        for (int nt = 0; nt < 4; ++nt) acc2[mi][nt] = vzero;

#pragma unroll
    for (int ks = 0; ks < 2; ++ks) {
        v16bf a0 = load_a_frag(sm_a, (wv * 2 + 0) * 16 + ml, ks, h);
        v16bf a1 = load_a_frag(sm_a, (wv * 2 + 1) * 16 + ml, ks, h);
#pragma unroll
        for (int nt = 0; nt < 4; ++nt) {
            v16bf bf = load_b_frag(wt_conv, nt * 16 + ml, ks, h);
            acc2[0][nt] = __builtin_amdgcn_wmma_f32_16x16x32_bf16(
                false, a0, false, bf, (short)0, acc2[0][nt], false, false);
            acc2[1][nt] = __builtin_amdgcn_wmma_f32_16x16x32_bf16(
                false, a1, false, bf, (short)0, acc2[1][nt], false, false);
        }
    }

    // ---- fused epilogue: y + bconv + x0*(1 + att + sa) -> yout [ch][272] ----
#pragma unroll
    for (int mi = 0; mi < 2; ++mi)
#pragma unroll
        for (int nt = 0; nt < 4; ++nt)
#pragma unroll
            for (int r = 0; r < 8; ++r) {
                const int p  = (wv * 2 + mi) * 16 + r + 8 * h;
                const int c  = nt * 16 + ml;
                const int pp = ((p >> 4) + 1) * 18 + (p & 15) + 1;
                const float x0v = sm_x0[c * 326 + pp];
                sm_big[c * 272 + p] = acc2[mi][nt][r] + sm_bc[c]
                                    + x0v * (1.f + sm_att[c] + sm_sa[p]);
            }
    __syncthreads();

    // ---- coalesced NCHW store ----
    {
        float* dst = outg + (size_t)b * 64 * 16384
                          + (size_t)(y0 + py) * 128 + (xb + px);
        for (int c = 0; c < 64; ++c)
            dst[(size_t)c * 16384] = sm_big[c * 272 + tid];
    }
}

// ---------------------------------------------------------------------------
extern "C" void kernel_launch(void* const* d_in, const int* in_sizes, int n_in,
                              void* d_out, int out_size, void* d_ws, size_t ws_size,
                              hipStream_t stream)
{
    const float* x0    = (const float*)d_in[0];
    const float* x1    = (const float*)d_in[1];
    const float* x2    = (const float*)d_in[2];
    const float* Wkc1  = (const float*)d_in[3];
    const float* Wkc2  = (const float*)d_in[4];
    const float* Wks1  = (const float*)d_in[5];
    const float* bks1  = (const float*)d_in[6];
    const float* Wks2  = (const float*)d_in[7];
    const float* bks2  = (const float*)d_in[8];
    const float* Wconv = (const float*)d_in[9];
    const float* bconv = (const float*)d_in[10];
    const float* Wca1  = (const float*)d_in[11];
    const float* Wca2  = (const float*)d_in[12];
    const float* Wsa   = (const float*)d_in[13];
    const float* bsa   = (const float*)d_in[14];
    float* out = (float*)d_out;

    char* ws = (char*)d_ws;
    float*  kcg  = (float*)ws;                        // 16*576 f32
    float*  attg = (float*)(ws + 16 * 576 * 4);       // 16*64  f32
    __bf16* wt1  = (__bf16*)(ws + 16 * 576 * 4 + 16 * 64 * 4);  // 4096 bf16
    __bf16* wt2  = wt1 + 4096;                        // 4096 bf16

    prep_small<<<16, 64, 0, stream>>>(x1, Wkc1, Wkc2, Wca1, Wca2, kcg, attg);
    pack_weights<<<32, 128, 0, stream>>>(Wks1, Wconv, wt1, wt2);

    dim3 grid(8, 8, 16);   // (W/16, H/16, B)
    ddf_main<<<grid, 256, SM_TOTAL, stream>>>(
        x0, x2, bks1, Wks2, bks2, bconv, Wsa, bsa,
        kcg, attg, wt1, wt2, out);
}
